// SelfAttentionClassical_65481071400472
// MI455X (gfx1250) — matrix operand order repaired and checked by reading, not compiled, not run on hardware
//
#include <hip/hip_runtime.h>
#include <cstdint>
#include <cstddef>

// ---------------------------------------------------------------------------
// Self-attention (N=8192, d=256), flash-attention style, bf16 WMMA on gfx1250.
//   Q = (X @ R) * (1/16)   (scale folded in)
//   K =  X @ E
//   out = softmax(Q K^T) @ X      (online softmax, no 8192x8192 materialized)
// K/V tiles double-buffered in LDS and staged with CDNA5 async global->LDS
// DMA (ASYNCcnt) so chunk c+1's staging overlaps chunk c's WMMA work.
// Workspace needed: ~17 MB  (Xb 4M | Qb 4M | Kb 4M | Vt 4M | Rt 128K | Et 128K)
// ---------------------------------------------------------------------------

typedef unsigned short u16;
typedef __attribute__((ext_vector_type(16))) __bf16 v16bf;
typedef __attribute__((ext_vector_type(8)))  float  v8f;

#define EMBED  256
#define NROWS  8192

// LDS row strides (elements) chosen for 16B alignment + conflict-free b128 reads
#define KSTR 264   // 528B rows: bank offset 4*r per row -> 16 distinct banksets
#define VSTR 40    // 80B rows:  bank offset 20*r mod 64 distinct for r=0..15
#define PSTR 40

union Frag {
  v16bf v;
  uint4 q[2];
};

__device__ __forceinline__ u16 f2bf(float f) {
  union { float f; uint32_t u; } x{f};
  uint32_t u = x.u + 0x7FFFu + ((x.u >> 16) & 1u);   // round-to-nearest-even
  return (u16)(u >> 16);
}

__device__ __forceinline__ v8f wmma_bf16(v16bf a, v16bf b, v8f c) {
  // D = A(16x32 bf16) x B(32x16 bf16) + C(16x16 f32)
  return __builtin_amdgcn_wmma_f32_16x16x32_bf16(false, a, false, b,
                                                 (short)0, c, false, false);
}

// Async DMA: global -> LDS, 16B per lane, tracked by ASYNCcnt (no VGPR data).
// VDST = 32-bit LDS byte address (low 32 bits of the generic LDS pointer per
// the aperture mapping), VADDR = 64-bit global address, SADDR = off (GV mode).
__device__ __forceinline__ void async_b128(const u16* g, const u16* l) {
  asm volatile("global_load_async_to_lds_b128 %0, %1, off"
               :: "v"((unsigned)(uintptr_t)l),
                  "v"((unsigned long long)(uintptr_t)g)
               : "memory");
}

// ---------------------------------------------------------------------------
// Phase 0a: X (f32, row-major [8192][256]) -> Xb (bf16 row-major)
//           and Vt (bf16, d-major [256][8192]) for contiguous V B-fragments.
__global__ void cvt_x_kernel(const float* __restrict__ X,
                             u16* __restrict__ Xb, u16* __restrict__ Vt) {
  int i = blockIdx.x * 256 + threadIdx.x;     // 8192*256 threads
  u16 b = f2bf(X[i]);
  Xb[i] = b;
  int r = i >> 8;        // row
  int d = i & 255;       // feature
  Vt[(size_t)d * NROWS + r] = b;
}

// Phase 0b: R,E (f32 [256][256]) -> transposed bf16 Rt,Et  (Rt[n][d] = R[d][n])
__global__ void cvt_w_kernel(const float* __restrict__ R,
                             const float* __restrict__ E,
                             u16* __restrict__ Rt, u16* __restrict__ Et) {
  int i = blockIdx.x * 256 + threadIdx.x;     // 65536 threads
  int d = i >> 8;
  int n = i & 255;
  Rt[n * EMBED + d] = f2bf(R[i]);
  Et[n * EMBED + d] = f2bf(E[i]);
}

// ---------------------------------------------------------------------------
// Phase 1: projection GEMM.  grid (512, 2), block 128 (4 waves).
// Block = 16 rows x 256 cols; each wave computes a 16x64 tile.
// blockIdx.y == 0 -> Qb = Xb@R * 1/16 ; == 1 -> Kb = Xb@E
__global__ __launch_bounds__(128) void proj_kernel(
    const u16* __restrict__ Xb, const u16* __restrict__ Rt,
    const u16* __restrict__ Et, u16* __restrict__ Qb, u16* __restrict__ Kb) {
  const int lane = threadIdx.x & 31;
  const int wave = threadIdx.x >> 5;
  const int hi = lane >> 4;     // half-wave select
  const int lm = lane & 15;
  const int row0 = blockIdx.x * 16;
  const int col0 = wave * 64;
  const u16* WT = blockIdx.y ? Et : Rt;
  u16* Out      = blockIdx.y ? Kb : Qb;
  const float scale = blockIdx.y ? 1.0f : 0.0625f;  // 1/sqrt(256) folded into Q

  const v8f zero = {0, 0, 0, 0, 0, 0, 0, 0};
  v8f c[4];
#pragma unroll
  for (int t = 0; t < 4; ++t) c[t] = zero;

#pragma unroll
  for (int dc = 0; dc < 8; ++dc) {
    // A-frag (16x32 bf16): lane holds row lm; 16 contiguous k at base+16*hi
    Frag a;
    const u16* xr = Xb + (row0 + lm) * EMBED + dc * 32 + hi * 16;
    a.q[0] = *(const uint4*)xr;
    a.q[1] = *(const uint4*)(xr + 8);
#pragma unroll
    for (int t = 0; t < 4; ++t) {
      // B-frag (32x16): lane = n + 16*(k>=16); 16 contiguous d values
      Frag b;
      const u16* wr = WT + (col0 + t * 16 + lm) * EMBED + dc * 32 + hi * 16;
      b.q[0] = *(const uint4*)wr;
      b.q[1] = *(const uint4*)(wr + 8);
      c[t] = wmma_bf16(a.v, b.v, c[t]);
    }
  }
  // D layout: VGPR r -> row (r + 8*hi), col = lm
#pragma unroll
  for (int t = 0; t < 4; ++t)
#pragma unroll
    for (int r = 0; r < 8; ++r)
      Out[(row0 + r + hi * 8) * EMBED + col0 + t * 16 + lm] =
          f2bf(c[t][r] * scale);
}

// ---------------------------------------------------------------------------
// Phase 2: flash attention. grid 128, block 128 (4 waves). Wave owns 16 rows.
__global__ __launch_bounds__(128, 1) void attn_kernel(
    const u16* __restrict__ Qb, const u16* __restrict__ Kb,
    const u16* __restrict__ Vt, float* __restrict__ out) {
  __shared__ u16 k_lds[2][32 * KSTR];       // K tile:  32 keys x 256 d  (x2)
  __shared__ u16 v_lds[2][256 * VSTR];      // Vt tile: 256 d x 32 keys  (x2)
  __shared__ u16 q_lds[4 * 16 * KSTR];      // per-wave Q: 16 rows x 256 d
  __shared__ u16 p_lds[4 * 16 * PSTR];      // per-wave P staging: 16 x 32

  const int lane = threadIdx.x & 31;
  const int wave = threadIdx.x >> 5;
  const int hi = lane >> 4;
  const int lm = lane & 15;
  const int qrow0 = blockIdx.x * 64 + wave * 16;

  // Issue async staging of one K/V chunk: 16 b128 DMAs per thread.
  auto stage = [&](int buf, int jb) {
    {
      int row = threadIdx.x >> 2;             // 0..31
      int seg = threadIdx.x & 3;              // 128B segment within row
      const u16* gk = Kb + (jb + row) * EMBED + seg * 64;
      const u16* lk = &k_lds[buf][row * KSTR + seg * 64];
#pragma unroll
      for (int i = 0; i < 8; ++i) async_b128(gk + i * 8, lk + i * 8);
    }
#pragma unroll
    for (int t = 0; t < 2; ++t) {
      int dd = threadIdx.x + t * 128;         // d-row of Vt
      const u16* gv = Vt + (size_t)dd * NROWS + jb;
      const u16* lv = &v_lds[buf][dd * VSTR];
#pragma unroll
      for (int i = 0; i < 4; ++i) async_b128(gv + i * 8, lv + i * 8);
    }
  };

  // stage this wave's Q block into LDS (2 lanes per row, 16 b128 each)
  {
    const uint4* src = (const uint4*)(Qb + (qrow0 + lm) * EMBED + hi * 128);
    uint4* dst = (uint4*)&q_lds[(wave * 16 + lm) * KSTR + hi * 128];
#pragma unroll
    for (int i = 0; i < 16; ++i) dst[i] = src[i];
  }

  const v8f zero = {0, 0, 0, 0, 0, 0, 0, 0};
  v8f O[16];
#pragma unroll
  for (int nt = 0; nt < 16; ++nt) O[nt] = zero;
  float mrow[8], lrow[8];
#pragma unroll
  for (int r = 0; r < 8; ++r) { mrow[r] = -3.0e38f; lrow[r] = 0.0f; }

  stage(0, 0);   // prologue: fill buffer 0

  for (int jb = 0; jb < NROWS; jb += 32) {
    const int cur = (jb >> 5) & 1;
    if (jb + 32 < NROWS) {
      stage(cur ^ 1, jb + 32);  // overlap next chunk's DMA with this compute
      // wait until only the 16 just-issued DMAs remain -> current tile ready
      asm volatile("s_wait_asynccnt 0x10" ::: "memory");
    } else {
      asm volatile("s_wait_asynccnt 0x0" ::: "memory");
    }
    __syncthreads();

    // --- S = Q K^T for two 16x16 tiles (keys jb..jb+15, jb+16..jb+31) ---
    v8f s0 = zero, s1 = zero;
#pragma unroll
    for (int dc = 0; dc < 8; ++dc) {
      Frag a;
      const u16* qr = &q_lds[(wave * 16 + lm) * KSTR + dc * 32 + hi * 16];
      a.q[0] = *(const uint4*)qr;
      a.q[1] = *(const uint4*)(qr + 8);
      Frag b0, b1;
      const u16* k0 = &k_lds[cur][lm * KSTR + dc * 32 + hi * 16];
      b0.q[0] = *(const uint4*)k0;
      b0.q[1] = *(const uint4*)(k0 + 8);
      const u16* k1 = &k_lds[cur][(16 + lm) * KSTR + dc * 32 + hi * 16];
      b1.q[0] = *(const uint4*)k1;
      b1.q[1] = *(const uint4*)(k1 + 8);
      s0 = wmma_bf16(a.v, b0.v, s0);
      s1 = wmma_bf16(a.v, b1.v, s1);
    }

    // --- online softmax. Row r+8*hi of S lives in VGPR r across 16 lanes ---
    float alpha[8];
#pragma unroll
    for (int r = 0; r < 8; ++r) {
      float v = fmaxf(s0[r], s1[r]);
      v = fmaxf(v, __shfl_xor(v, 1, 32));
      v = fmaxf(v, __shfl_xor(v, 2, 32));
      v = fmaxf(v, __shfl_xor(v, 4, 32));
      v = fmaxf(v, __shfl_xor(v, 8, 32));
      float nm = fmaxf(mrow[r], v);
      alpha[r] = __expf(mrow[r] - nm);
      mrow[r] = nm;
    }
    float p0a[8], p1a[8];
#pragma unroll
    for (int r = 0; r < 8; ++r) {
      p0a[r] = __expf(s0[r] - mrow[r]);
      p1a[r] = __expf(s1[r] - mrow[r]);
      float rs = p0a[r] + p1a[r];
      rs += __shfl_xor(rs, 1, 32);
      rs += __shfl_xor(rs, 2, 32);
      rs += __shfl_xor(rs, 4, 32);
      rs += __shfl_xor(rs, 8, 32);
      lrow[r] = lrow[r] * alpha[r] + rs;
    }
    // rescale accumulators
#pragma unroll
    for (int nt = 0; nt < 16; ++nt)
#pragma unroll
      for (int r = 0; r < 8; ++r) O[nt][r] *= alpha[r];

    // --- stage P (D layout -> A layout) through per-wave LDS ---
    {
      u16* pp = &p_lds[wave * 16 * PSTR];
#pragma unroll
      for (int r = 0; r < 8; ++r) {
        int row = r + hi * 8;
        pp[row * PSTR + lm]      = f2bf(p0a[r]);
        pp[row * PSTR + 16 + lm] = f2bf(p1a[r]);
      }
    }
    asm volatile("s_wait_dscnt 0" ::: "memory");
    Frag pa;
    {
      const u16* pr = &p_lds[(wave * 16 + lm) * PSTR + hi * 16];
      pa.q[0] = *(const uint4*)pr;
      pa.q[1] = *(const uint4*)(pr + 8);
    }
    // --- O += P V  (V = X, read from Vt tile in LDS) ---
#pragma unroll
    for (int nt = 0; nt < 16; ++nt) {
      Frag vb;
      const u16* vr = &v_lds[cur][(nt * 16 + lm) * VSTR + hi * 16];
      vb.q[0] = *(const uint4*)vr;
      vb.q[1] = *(const uint4*)(vr + 8);
      O[nt] = wmma_bf16(pa.v, vb.v, O[nt]);
    }
    __syncthreads();   // all waves done reading buf[cur] -> safe to re-stage
  }

  // epilogue: normalize rows and store f32
  float inv[8];
#pragma unroll
  for (int r = 0; r < 8; ++r) inv[r] = 1.0f / lrow[r];
#pragma unroll
  for (int nt = 0; nt < 16; ++nt)
#pragma unroll
    for (int r = 0; r < 8; ++r)
      out[(qrow0 + r + hi * 8) * EMBED + nt * 16 + lm] = O[nt][r] * inv[r];
}

// ---------------------------------------------------------------------------
extern "C" void kernel_launch(void* const* d_in, const int* in_sizes, int n_in,
                              void* d_out, int out_size, void* d_ws,
                              size_t ws_size, hipStream_t stream) {
  const float* R = (const float*)d_in[0];   // rotation_params [256,256]
  const float* E = (const float*)d_in[1];   // entangle_params [256,256]
  const float* X = (const float*)d_in[2];   // inputs [8192,256]
  float* out = (float*)d_out;

  char* ws = (char*)d_ws;
  u16* Xb = (u16*)(ws);                               // 4 MB
  u16* Qb = (u16*)(ws + ((size_t)4 << 20));           // 4 MB
  u16* Kb = (u16*)(ws + ((size_t)8 << 20));           // 4 MB
  u16* Vt = (u16*)(ws + ((size_t)12 << 20));          // 4 MB
  u16* Rt = (u16*)(ws + ((size_t)16 << 20));          // 128 KB
  u16* Et = (u16*)(ws + ((size_t)16 << 20) + (1 << 17));

  cvt_x_kernel<<<NROWS, 256, 0, stream>>>(X, Xb, Vt);
  cvt_w_kernel<<<256, 256, 0, stream>>>(R, E, Rt, Et);
  proj_kernel<<<dim3(NROWS / 16, 2), 128, 0, stream>>>(Xb, Rt, Et, Qb, Kb);
  attn_kernel<<<NROWS / 64, 128, 0, stream>>>(Qb, Kb, Vt, out);
}
